// SelectiveSSM_66735201845459
// MI455X (gfx1250) — matrix operand order repaired and checked by reading, not compiled, not run on hardware
//
#include <hip/hip_runtime.h>
#include <cstddef>

// ---------------------------------------------------------------------------
// Mamba selective-SSM forward for gfx1250 (MI455X), wave32 + WMMA f16.
// B=2, L=2048, D_MODEL=1024, D_INNER=2048, D_STATE=16, D_CONV=4, DT_RANK=64
// ---------------------------------------------------------------------------

constexpr int kB      = 2;
constexpr int kL      = 2048;
constexpr int kDM     = 1024;
constexpr int kDI     = 2048;
constexpr int kDS     = 16;
constexpr int kDConv  = 4;
constexpr int kDtRank = 64;
constexpr int kRows   = kB * kL;              // 4096 token rows
constexpr int kNX     = kDtRank + 2 * kDS;    // 96
constexpr int kNCH    = 32;                   // scan chunks
constexpr int kCLEN   = 64;                   // steps per chunk (32*64 = L)

typedef __attribute__((ext_vector_type(16))) _Float16 v16h;
typedef __attribute__((ext_vector_type(8)))  _Float16 v8h;
typedef __attribute__((ext_vector_type(8)))  float    v8f;

__device__ __forceinline__ float silu_(float x)  { return x / (1.0f + __expf(-x)); }
__device__ __forceinline__ float softplus_(float x) {
    return (x > 20.0f) ? x : log1pf(__expf(x));
}

// ---------------------------------------------------------------------------
// Weight prep: W[K][N] fp32  ->  Wt[N][K] f16   (LDS-tiled 32x32 transpose)
// K, N multiples of 32.
// ---------------------------------------------------------------------------
__global__ void __launch_bounds__(256)
transpose_cvt_f16(const float* __restrict__ W, _Float16* __restrict__ Wt,
                  int K, int N)
{
    __shared__ float tile[32][33];
    const int tpr = N >> 5;
    const int tn0 = (blockIdx.x % tpr) << 5;
    const int tk0 = (blockIdx.x / tpr) << 5;
    const int tx  = threadIdx.x & 31;
    const int ty  = threadIdx.x >> 5;
    #pragma unroll
    for (int r = 0; r < 4; ++r) {
        const int kk = ty + r * 8;
        tile[kk][tx] = W[(size_t)(tk0 + kk) * N + tn0 + tx];
    }
    __syncthreads();
    #pragma unroll
    for (int r = 0; r < 4; ++r) {
        const int nn = ty + r * 8;
        Wt[(size_t)(tn0 + nn) * K + tk0 + tx] = (_Float16)tile[tx][nn];
    }
}

// Flat fp32 -> f16 convert (activations)
__global__ void __launch_bounds__(256)
cvt_f16_kernel(const float* __restrict__ in, _Float16* __restrict__ out, size_t n)
{
    const size_t i = (size_t)blockIdx.x * 256 + threadIdx.x;
    if (i < n) out[i] = (_Float16)in[i];
}

// ---------------------------------------------------------------------------
// GEMM: C[M,N](f32) = A[M,K](f16, row-major) * Bt[N,K](f16, i.e. B transposed)
// One wave computes a 64x32 tile: 4x2 WMMA tiles, 8 v_wmma per 32-wide K-step.
// All operand fetches are contiguous 16B -> global_load_b128.
// Requires M%64==0, N%32==0, K%32==0 (true for every call here).
// ---------------------------------------------------------------------------
__global__ void __launch_bounds__(256)
gemm_wmma_f16t(const _Float16* __restrict__ A, const _Float16* __restrict__ Bt,
               float* __restrict__ C, int M, int N, int K,
               int lda, int ldb, int ldc)
{
    const int lane = threadIdx.x & 31;
    const int wave = threadIdx.x >> 5;
    const int ntn  = N >> 5;
    const int ntot = (M >> 6) * ntn;
    const int tile = blockIdx.x * 8 + wave;
    if (tile >= ntot) return;                 // wave-uniform (EXEC stays full)

    const int tm   = (tile / ntn) << 6;
    const int tn   = (tile % ntn) << 5;
    const int mrow = lane & 15;
    const int hi   = lane >> 4;

    // A fragment element j -> k = k0 + 16*hi + j
    const _Float16* aBase[4];
    #pragma unroll
    for (int mt = 0; mt < 4; ++mt)
        aBase[mt] = A + (size_t)(tm + mt * 16 + mrow) * lda + hi * 16;
    // B fragment element j -> k = k0 + 16*hi + j, at fixed n
    const _Float16* bBase[2];
    #pragma unroll
    for (int nt = 0; nt < 2; ++nt)
        bBase[nt] = Bt + (size_t)(tn + nt * 16 + mrow) * ldb + hi * 16;

    v8f acc[4][2] = {};

    for (int k0 = 0; k0 < K; k0 += 32) {
        v16h af[4], bf[2];
        #pragma unroll
        for (int mt = 0; mt < 4; ++mt) {
            const v8h lo  = *(const v8h*)(aBase[mt] + k0);
            const v8h hi8 = *(const v8h*)(aBase[mt] + k0 + 8);
            af[mt] = __builtin_shufflevector(lo, hi8, 0,1,2,3,4,5,6,7,
                                                      8,9,10,11,12,13,14,15);
        }
        #pragma unroll
        for (int nt = 0; nt < 2; ++nt) {
            const v8h lo  = *(const v8h*)(bBase[nt] + k0);
            const v8h hi8 = *(const v8h*)(bBase[nt] + k0 + 8);
            bf[nt] = __builtin_shufflevector(lo, hi8, 0,1,2,3,4,5,6,7,
                                                      8,9,10,11,12,13,14,15);
        }
        if (k0 + 32 < K) {                    // uniform; next K-slice prefetch
            __builtin_prefetch(aBase[0] + k0 + 32, 0, 1);
            __builtin_prefetch(bBase[0] + k0 + 32, 0, 1);
        }
        #pragma unroll
        for (int mt = 0; mt < 4; ++mt)
            #pragma unroll
            for (int nt = 0; nt < 2; ++nt)
                acc[mt][nt] = __builtin_amdgcn_wmma_f32_16x16x32_f16(
                    false, af[mt], false, bf[nt],
                    (short)0, acc[mt][nt], false, false);
    }

    // D layout: vgpr r -> (m = r + 8*hi, n = lane&15)
    #pragma unroll
    for (int mt = 0; mt < 4; ++mt)
        #pragma unroll
        for (int nt = 0; nt < 2; ++nt)
            #pragma unroll
            for (int r = 0; r < 8; ++r)
                C[(size_t)(tm + mt * 16 + r + hi * 8) * ldc +
                  tn + nt * 16 + mrow] = acc[mt][nt][r];
}

// ---------------------------------------------------------------------------
// Depthwise causal conv1d (k=4) + bias + SiLU -> fp32 (for scan) + f16 (GEMM A)
// ---------------------------------------------------------------------------
__global__ void __launch_bounds__(256)
conv_silu_kernel(const float* __restrict__ xz, const float* __restrict__ cw,
                 const float* __restrict__ cb, float* __restrict__ xs32,
                 _Float16* __restrict__ xs16)
{
    const size_t idx = (size_t)blockIdx.x * 256 + threadIdx.x;
    if (idx >= (size_t)kRows * kDI) return;
    const int d  = (int)(idx % kDI);
    const int bl = (int)(idx / kDI);
    const int l  = bl % kL;

    float acc = cb[d];
    #pragma unroll
    for (int j = 0; j < kDConv; ++j) {
        const int ll = l - (kDConv - 1) + j;
        if (ll >= 0)
            acc += xz[(size_t)(bl - (kDConv - 1) + j) * (2 * kDI) + d]
                   * cw[d * kDConv + j];
    }
    const float v = silu_(acc);
    xs32[idx] = v;
    xs16[idx] = (_Float16)v;
}

// ---------------------------------------------------------------------------
// Chunked selective scan (affine recurrence => chunk transition is
// exp(A * sum(dt)) since dA_t = exp(A*dt_t)).
//   S1: per-chunk local recurrence from h=0, record h_end[16] and sum(dt)
//   S2: serial combine of 32 chunk summaries per channel -> h0 per chunk
//   S3: replay each chunk from its true h0, produce gated y in f16
// ---------------------------------------------------------------------------
__global__ void __launch_bounds__(256)
scan_chunk_local(const float* __restrict__ xs, const float* __restrict__ draw,
                 const float* __restrict__ x_dbl, const float* __restrict__ A_log,
                 const float* __restrict__ dt_bias,
                 float* __restrict__ hend, float* __restrict__ sdt)
{
    __shared__ float sB[kCLEN][kDS];
    const int c    = blockIdx.x & (kNCH - 1);
    const int rem  = blockIdx.x >> 5;            // kNCH == 32
    const int b    = rem >> 3;                   // kDI/256 == 8
    const int d    = ((rem & 7) << 8) + threadIdx.x;
    const int row0 = b * kL + c * kCLEN;

    #pragma unroll
    for (int i = 0; i < (kCLEN * kDS) / 256; ++i) {   // 4 x 256 = 1024 floats
        const int idx = threadIdx.x + i * 256;
        sB[idx >> 4][idx & 15] =
            x_dbl[(size_t)(row0 + (idx >> 4)) * kNX + kDtRank + (idx & 15)];
    }
    __syncthreads();

    float Ac[kDS], h[kDS];
    #pragma unroll
    for (int s = 0; s < kDS; ++s) { Ac[s] = -__expf(A_log[d * kDS + s]); h[s] = 0.f; }
    const float dbias = dt_bias[d];
    float S = 0.0f;

    for (int tt = 0; tt < kCLEN; ++tt) {
        const size_t row = row0 + tt;
        const float u  = xs[row * kDI + d];
        const float dt = softplus_(draw[row * kDI + d] + dbias);
        const float du = dt * u;
        S += dt;
        #pragma unroll
        for (int s = 0; s < kDS; ++s)
            h[s] = __expf(dt * Ac[s]) * h[s] + du * sB[tt][s];
    }
    const size_t ch = ((size_t)(b * kDI + d) * kNCH + c);
    #pragma unroll
    for (int s = 0; s < kDS; ++s) hend[ch * kDS + s] = h[s];
    sdt[ch] = S;
}

__global__ void __launch_bounds__(256)
scan_combine(const float* __restrict__ A_log, const float* __restrict__ hend,
             const float* __restrict__ sdt, float* __restrict__ h0s)
{
    const int chn = blockIdx.x * 256 + threadIdx.x;   // 0..4095
    const int d   = chn & (kDI - 1);
    float Ac[kDS], h[kDS];
    #pragma unroll
    for (int s = 0; s < kDS; ++s) { Ac[s] = -__expf(A_log[d * kDS + s]); h[s] = 0.f; }
    for (int c = 0; c < kNCH; ++c) {
        const size_t base = ((size_t)chn * kNCH + c) * kDS;
        const float  S    = sdt[(size_t)chn * kNCH + c];
        #pragma unroll
        for (int s = 0; s < kDS; ++s) {
            h0s[base + s] = h[s];
            h[s] = __expf(Ac[s] * S) * h[s] + hend[base + s];
        }
    }
}

__global__ void __launch_bounds__(256)
scan_chunk_final(const float* __restrict__ xz, const float* __restrict__ xs,
                 const float* __restrict__ draw, const float* __restrict__ x_dbl,
                 const float* __restrict__ A_log, const float* __restrict__ dt_bias,
                 const float* __restrict__ D_param, const float* __restrict__ h0s,
                 _Float16* __restrict__ y16)
{
    __shared__ float sB[kCLEN][kDS];
    __shared__ float sC[kCLEN][kDS];
    const int c    = blockIdx.x & (kNCH - 1);
    const int rem  = blockIdx.x >> 5;
    const int b    = rem >> 3;
    const int d    = ((rem & 7) << 8) + threadIdx.x;
    const int row0 = b * kL + c * kCLEN;

    #pragma unroll
    for (int i = 0; i < (kCLEN * 2 * kDS) / 256; ++i) {   // 8 x 256 = 2048 floats
        const int idx = threadIdx.x + i * 256;
        const int tt  = idx >> 5;
        const int j   = idx & 31;
        const float v = x_dbl[(size_t)(row0 + tt) * kNX + kDtRank + j];
        if (j < kDS) sB[tt][j] = v; else sC[tt][j - kDS] = v;
    }
    __syncthreads();

    float Ac[kDS], h[kDS];
    const size_t ch = ((size_t)(b * kDI + d) * kNCH + c);
    #pragma unroll
    for (int s = 0; s < kDS; ++s) {
        Ac[s] = -__expf(A_log[d * kDS + s]);
        h[s]  = h0s[ch * kDS + s];
    }
    const float dbias = dt_bias[d];
    const float Dv    = D_param[d];

    for (int tt = 0; tt < kCLEN; ++tt) {
        const size_t row = row0 + tt;
        const float u  = xs[row * kDI + d];
        const float dt = softplus_(draw[row * kDI + d] + dbias);
        const float du = dt * u;
        float y = 0.0f;
        #pragma unroll
        for (int s = 0; s < kDS; ++s) {
            h[s] = __expf(dt * Ac[s]) * h[s] + du * sB[tt][s];
            y   += h[s] * sC[tt][s];
        }
        const float z = xz[row * (2 * kDI) + kDI + d];
        y16[row * kDI + d] = (_Float16)((y + u * Dv) * silu_(z));
    }
}

// ---------------------------------------------------------------------------
// Host-side launch
// ---------------------------------------------------------------------------
extern "C" void kernel_launch(void* const* d_in, const int* in_sizes, int n_in,
                              void* d_out, int out_size, void* d_ws, size_t ws_size,
                              hipStream_t stream)
{
    const float* x         = (const float*)d_in[0];
    const float* in_proj_w = (const float*)d_in[1];
    const float* conv_w    = (const float*)d_in[2];
    const float* conv_b    = (const float*)d_in[3];
    const float* x_proj_w  = (const float*)d_in[4];
    const float* dt_proj_w = (const float*)d_in[5];
    const float* dt_proj_b = (const float*)d_in[6];
    const float* A_log     = (const float*)d_in[7];
    const float* D_param   = (const float*)d_in[8];
    const float* out_proj  = (const float*)d_in[9];
    float* out = (float*)d_out;

    // ---- workspace carve-out -------------------------------------------
    const size_t n_xz   = (size_t)kRows * (2 * kDI);
    const size_t n_xs   = (size_t)kRows * kDI;
    const size_t n_drw  = (size_t)kRows * kDI;
    const size_t n_xdbl = (size_t)kRows * kNX;
    const size_t n_hend = (size_t)kB * kDI * kNCH * kDS;
    const size_t n_h0s  = n_hend;
    const size_t n_sdt  = (size_t)kB * kDI * kNCH;
    const size_t nfloat = n_xz + n_xs + n_drw + n_xdbl + n_hend + n_h0s + n_sdt;

    const size_t h_x16  = (size_t)kRows * kDM;
    const size_t h_xs16 = (size_t)kRows * kDI;
    const size_t h_xd16 = (size_t)kRows * kNX;
    const size_t h_wi   = (size_t)(2 * kDI) * kDM;   // in_proj^T
    const size_t h_wxp  = (size_t)kNX * kDI;         // x_proj^T
    const size_t h_wdt  = (size_t)kDI * kDtRank;     // dt_proj^T
    const size_t h_wo   = (size_t)kDM * kDI;         // out_proj^T
    const size_t nhalf  = h_x16 + h_xs16 + h_xd16 + h_wi + h_wxp + h_wdt + h_wo;

    if (ws_size < nfloat * sizeof(float) + nhalf * sizeof(_Float16)) return;

    float* xz        = (float*)d_ws;
    float* xs32      = xz + n_xz;
    float* delta_raw = xs32 + n_xs;
    float* x_dbl     = delta_raw + n_drw;
    float* hend      = x_dbl + n_xdbl;
    float* h0s       = hend + n_hend;
    float* sdt       = h0s + n_h0s;

    _Float16* x16    = (_Float16*)(sdt + n_sdt);
    _Float16* xs16   = x16 + h_x16;
    _Float16* xd16   = xs16 + h_xs16;
    _Float16* w_in_t = xd16 + h_xd16;
    _Float16* w_xp_t = w_in_t + h_wi;
    _Float16* w_dt_t = w_xp_t + h_wxp;
    _Float16* w_o_t  = w_dt_t + h_wdt;

    // ---- 0) one-time precision prep ------------------------------------
    transpose_cvt_f16<<<(kDM / 32) * ((2 * kDI) / 32), 256, 0, stream>>>(
        in_proj_w, w_in_t, kDM, 2 * kDI);
    transpose_cvt_f16<<<(kDI / 32) * (kNX / 32), 256, 0, stream>>>(
        x_proj_w, w_xp_t, kDI, kNX);
    transpose_cvt_f16<<<(kDtRank / 32) * (kDI / 32), 256, 0, stream>>>(
        dt_proj_w, w_dt_t, kDtRank, kDI);
    transpose_cvt_f16<<<(kDI / 32) * (kDM / 32), 256, 0, stream>>>(
        out_proj, w_o_t, kDI, kDM);
    cvt_f16_kernel<<<(int)((h_x16 + 255) / 256), 256, 0, stream>>>(x, x16, h_x16);

    // ---- 1) xz = x @ in_proj_w      (4096x1024 @ 1024x4096) ------------
    gemm_wmma_f16t<<<(kRows / 64) * ((2 * kDI) / 32) / 8, 256, 0, stream>>>(
        x16, w_in_t, xz, kRows, 2 * kDI, kDM, kDM, kDM, 2 * kDI);
    // ---- 2) depthwise conv + SiLU --------------------------------------
    conv_silu_kernel<<<(int)(((size_t)kRows * kDI + 255) / 256), 256, 0, stream>>>(
        xz, conv_w, conv_b, xs32, xs16);
    // ---- 3) x_dbl = xs @ x_proj_w   (4096x2048 @ 2048x96) --------------
    gemm_wmma_f16t<<<(kRows / 64) * (kNX / 32) / 8, 256, 0, stream>>>(
        xs16, w_xp_t, x_dbl, kRows, kNX, kDI, kDI, kDI, kNX);
    cvt_f16_kernel<<<(int)((h_xd16 + 255) / 256), 256, 0, stream>>>(x_dbl, xd16, h_xd16);
    // ---- 4) delta_raw = dlt @ dt_proj_w  (4096x64 @ 64x2048) -----------
    gemm_wmma_f16t<<<(kRows / 64) * (kDI / 32) / 8, 256, 0, stream>>>(
        xd16, w_dt_t, delta_raw, kRows, kDI, kDtRank, kNX, kDtRank, kDI);
    // ---- 5) chunked selective scan -------------------------------------
    scan_chunk_local<<<kB * (kDI / 256) * kNCH, 256, 0, stream>>>(
        xs32, delta_raw, x_dbl, A_log, dt_proj_b, hend, sdt);
    scan_combine<<<kB * kDI / 256, 256, 0, stream>>>(A_log, hend, sdt, h0s);
    scan_chunk_final<<<kB * (kDI / 256) * kNCH, 256, 0, stream>>>(
        xz, xs32, delta_raw, x_dbl, A_log, dt_proj_b, D_param, h0s, xs16 /*y16*/);
    // ---- 6) out = y @ out_proj_w    (4096x2048 @ 2048x1024) ------------
    gemm_wmma_f16t<<<(kRows / 64) * (kDM / 32) / 8, 256, 0, stream>>>(
        xs16, w_o_t, out, kRows, kDM, kDI, kDI, kDI, kDM);
}
